// CausalAttention_56891136803058
// MI455X (gfx1250) — matrix operand order repaired and checked by reading, not compiled, not run on hardware
//
#include <hip/hip_runtime.h>
#include <hip/hip_bf16.h>
#include <math.h>
#include <stdint.h>

// ---------------------------------------------------------------------------
// CausalAttention for MI455X (gfx1250, wave32, WMMA f16 -> f32, TDM staging)
// B=4, N=4096, D=1024.  y = softmax(mask(QK^T/32)) V,  Q=xWq^T, K=xWk^T, V=xWv^T
// ---------------------------------------------------------------------------

typedef __attribute__((ext_vector_type(16))) _Float16 v16h;
typedef __attribute__((ext_vector_type(8)))  _Float16 v8h;
typedef __attribute__((ext_vector_type(8)))  float    v8f;
typedef __attribute__((ext_vector_type(4)))  unsigned int v4u;
typedef __attribute__((ext_vector_type(8)))  int      v8i;
typedef __attribute__((ext_vector_type(4)))  int      v4i;

constexpr int Bv = 4;
constexpr int Nv = 4096;
constexpr int Dv = 1024;
constexpr float SM_SCALE = 0.03125f;   // 1/sqrt(1024)

// LDS row strides (elements), chosen so every vector access is 16B-aligned
constexpr int SQW = 1032;   // f16, 1032*2 = 2064 = 16*129
constexpr int SW  = 132;    // f32 score tile stride
constexpr int SPW = 136;    // f16, 136*2 = 272 = 16*17

static __device__ __forceinline__ v8f wmma_f16(const v16h& a, const v16h& b, const v8f& c) {
    return __builtin_amdgcn_wmma_f32_16x16x32_f16(false, a, false, b, (short)0, c, false, false);
}

// ---------------------------------------------------------------------------
// Kernel 0: one-shot f32 -> f16 weight conversion (Wq|Wk|Wv concatenated).
// 6MB of f16 weights stay hot in L2 for the whole QKV pass.
// ---------------------------------------------------------------------------
__global__ void __launch_bounds__(256)
convert_w_kernel(const float* __restrict__ Wq,
                 const float* __restrict__ Wk,
                 const float* __restrict__ Wv,
                 _Float16* __restrict__ Wh)
{
    const float* src = (blockIdx.y == 0) ? Wq : (blockIdx.y == 1) ? Wk : Wv;
    _Float16* dst = Wh + (size_t)blockIdx.y * Dv * Dv;
    const size_t i = ((size_t)blockIdx.x * 256 + threadIdx.x) * 8;
    #pragma unroll
    for (int e = 0; e < 8; ++e)
        dst[i + e] = (_Float16)src[i + e];
}

// ---------------------------------------------------------------------------
// Kernel 1: QKV projection.  grid = (M/64, Dout/128, 3), block = 128 (4 waves).
// Each wave computes a 16(M) x 128(O) strip: 8 WMMAs per K-step, A reused 8x.
// z=0 -> Qh (f16 row-major), z=1 -> Kh (f16 row-major), z=2 -> Vt (f16, [b][d][n])
// ---------------------------------------------------------------------------
__global__ void __launch_bounds__(128)
qkv_gemm_kernel(const float* __restrict__ x,
                const _Float16* __restrict__ Wh,
                _Float16* __restrict__ Qh,
                _Float16* __restrict__ Kh,
                _Float16* __restrict__ Vt)
{
    const int lane = threadIdx.x & 31;
    const int wave = threadIdx.x >> 5;          // 0..3
    const int ln   = lane & 15;
    const int half = lane >> 4;                 // K-half select (A/B layouts)
    const int m0   = blockIdx.x * 64 + wave * 16;   // token-row tile (over B*N)
    const int o0   = blockIdx.y * 128;              // output-feature tile
    const int z    = blockIdx.z;                    // 0=Q 1=K 2=V
    const _Float16* W = Wh + (size_t)z * Dv * Dv;

    v8f acc[8] = {};
    const float* xrow = x + (size_t)(m0 + ln) * Dv;

    for (int kk = 0; kk < Dv; kk += 32) {
        // A fragment: contiguous k order, same as B.  elems 0..7 = K kk+half*16..,
        // elems 8..15 = K kk+half*16+8..  (fused f32->f16 convert)
        v16h a;
        {
            const float* p0 = xrow + kk + half * 16;
            const float* p1 = xrow + kk + half * 16 + 8;
            #pragma unroll
            for (int e = 0; e < 8; ++e) {
                a[e]     = (_Float16)p0[e];
                a[8 + e] = (_Float16)p1[e];
            }
        }
        #pragma unroll
        for (int t = 0; t < 8; ++t) {
            // B fragment: B[k][n] = W[n][k]; lane holds column n = o0+t*16+ln,
            // K range half*16..half*16+15 -> contiguous 16-half read of a W row.
            const v16h b = *(const v16h*)(W + (size_t)(o0 + t * 16 + ln) * Dv + kk + half * 16);
            acc[t] = wmma_f16(a, b, acc[t]);
        }
    }

    if (z < 2) {
        _Float16* out = (z == 0) ? Qh : Kh;
        #pragma unroll
        for (int t = 0; t < 8; ++t)
            #pragma unroll
            for (int r = 0; r < 8; ++r) {
                const int row = m0 + r + 8 * half;       // C layout: VGPR r -> M=r+8*half
                const int col = o0 + t * 16 + ln;
                out[(size_t)row * Dv + col] = (_Float16)acc[t][r];
            }
    } else {
        // V stored transposed: Vt[b][d][n] so attention B-fragments are contiguous
        #pragma unroll
        for (int t = 0; t < 8; ++t)
            #pragma unroll
            for (int r = 0; r < 8; ++r) {
                const int row = m0 + r + 8 * half;       // global token = b*N + n
                const int bb  = row >> 12;               // / 4096
                const int n   = row & (Nv - 1);
                const int col = o0 + t * 16 + ln;
                Vt[(size_t)bb * Dv * Nv + (size_t)col * Nv + n] = (_Float16)acc[t][r];
            }
    }
}

// ---------------------------------------------------------------------------
// Kernel 2: flash attention.  grid = (N/16, B), block = 256 (8 waves).
// Q tile (16x1024 f16, padded rows) staged into LDS by the Tensor Data Mover
// (iterate mode: 1 row per iteration, lds_addr_increment = padded stride).
// ---------------------------------------------------------------------------
__global__ void __launch_bounds__(256)
attn_kernel(const _Float16* __restrict__ Qh,
            const _Float16* __restrict__ Kh,
            const _Float16* __restrict__ Vt,
            float* __restrict__ out)
{
    __shared__ _Float16 sQ[16 * SQW];
    __shared__ float    sS[16 * SW];
    __shared__ _Float16 sP[16 * SPW];
    __shared__ float    m_s[16], l_s[16], scale_s[16];

    const int tid  = threadIdx.x;
    const int lane = tid & 31;
    const int wave = tid >> 5;        // 0..7
    const int ln   = lane & 15;
    const int half = lane >> 4;
    const int q0   = blockIdx.x * 16; // query tile start (within batch)
    const int bb   = blockIdx.y;

    if (tid < 16) { m_s[tid] = -INFINITY; l_s[tid] = 0.0f; }

#if __has_builtin(__builtin_amdgcn_tensor_load_to_lds)
    // ---- TDM: DMA the 16x1024 Q tile into LDS with padded row stride ------
    if (wave == 0) {
        const unsigned long long gaddr =
            (unsigned long long)(uintptr_t)(Qh + ((size_t)bb * Nv + q0) * Dv);
        const unsigned int lds_off = (unsigned int)(uintptr_t)(&sQ[0]);
        v4u g0;
        g0[0] = 1u;                                        // count=1 (valid user D#)
        g0[1] = lds_off;                                   // lds_addr (bytes)
        g0[2] = (unsigned int)(gaddr & 0xFFFFFFFFu);       // global_addr[31:0]
        g0[3] = (unsigned int)((gaddr >> 32) & 0x1FFFFFFu) // global_addr[56:32]
              | (2u << 30);                                // type=2 ("image")
        v8i g1 = {};
        g1[0] = (1 << 16) | (1 << 19);     // data_size=2B, iterate_enable=1
        g1[1] = (int)(1024u << 16);        // tensor_dim0[15:0]=1024 (bits 63:48)
        g1[2] = (int)(16u << 16);          // tensor_dim1[15:0]=16   (bits 95:80)
        g1[3] = (int)(1024u << 16);        // tile_dim0=1024         (bits 127:112)
        g1[4] = 1;                         // tile_dim1=1, tile_dim2=0
        g1[5] = 1024;                      // tensor_dim0_stride[31:0]
        g1[6] = (int)(1024u << 16);        // tensor_dim1_stride[15:0]
        g1[7] = 0;
        v4i g2;
        g2[0] = 0;                         // tensor_dim2 (unused in iterate mode)
        g2[1] = SQW;                       // lds_addr_increment (data_size units)
        g2[2] = Dv;                        // global_addr_increment (data_size units)
        g2[3] = (int)(15u << 16);          // iterate_count=15 -> 16 rows
        v4i g3 = {};
#if __clang_major__ >= 23
        v8i g4 = {};
        __builtin_amdgcn_tensor_load_to_lds(g0, g1, g2, g3, g4, 0);
#else
        __builtin_amdgcn_tensor_load_to_lds(g0, g1, g2, g3, 0);
#endif
        __builtin_amdgcn_s_wait_tensorcnt(0);
    }
#else
    // ---- fallback: manual cooperative copy --------------------------------
    {
        const int row = tid >> 4;
        const int c16 = tid & 15;
        const _Float16* src = Qh + ((size_t)bb * Nv + q0 + row) * Dv + c16 * 64;
        _Float16* dst = sQ + row * SQW + c16 * 64;
        #pragma unroll
        for (int k = 0; k < 8; ++k)
            *(v8h*)(dst + k * 8) = *(const v8h*)(src + k * 8);
    }
#endif
    __syncthreads();

    v8f oacc[8] = {};
    const int dbase = wave * 128;
    const int jend  = q0 + 15;

    for (int j0 = 0; j0 <= jend; j0 += 128) {
        // -------- phase 1: S = Q K^T (this wave's 16-key subtile) ----------
        {
            if (j0 + 128 <= jend) // warm next key block into cache
                __builtin_prefetch(Kh + ((size_t)bb * Nv + j0 + 128 + wave * 16 + ln) * Dv, 0, 1);

            const int jt = j0 + wave * 16;
            const _Float16* krow = Kh + ((size_t)bb * Nv + jt + ln) * Dv;
            v8f s = {};
            for (int kk = 0; kk < Dv; kk += 32) {
                const v8h a0 = *(const v8h*)(sQ + ln * SQW + kk + half * 16);
                const v8h a1 = *(const v8h*)(sQ + ln * SQW + kk + half * 16 + 8);
                v16h a;
                #pragma unroll
                for (int e = 0; e < 8; ++e) { a[e] = a0[e]; a[8 + e] = a1[e]; }
                const v16h b = *(const v16h*)(krow + kk + half * 16);
                s = wmma_f16(a, b, s);
            }
            #pragma unroll
            for (int r = 0; r < 8; ++r)
                sS[(r + 8 * half) * SW + wave * 16 + ln] = s[r];
        }
        __syncthreads();

        // -------- online softmax update (16 threads per query row) --------
        {
            const int row = tid >> 4;
            const int c   = tid & 15;
            const int gi  = q0 + row;
            float vals[8];
            float tmax = -INFINITY;
            #pragma unroll
            for (int e = 0; e < 8; ++e) {
                const int col = c * 8 + e;
                const float sv = sS[row * SW + col] * SM_SCALE;
                vals[e] = ((j0 + col) <= gi) ? sv : -INFINITY;
                tmax = fmaxf(tmax, vals[e]);
            }
            #pragma unroll
            for (int msk = 1; msk < 16; msk <<= 1)   // masks <=8 stay within 16-lane row group
                tmax = fmaxf(tmax, __shfl_xor(tmax, msk, 32));
            const float mold = m_s[row];
            const float mnew = fmaxf(mold, tmax);
            float sum = 0.0f;
            #pragma unroll
            for (int e = 0; e < 8; ++e) {
                const float p = (vals[e] == -INFINITY) ? 0.0f : __expf(vals[e] - mnew);
                sum += p;
                sP[row * SPW + c * 8 + e] = (_Float16)p;
            }
            #pragma unroll
            for (int msk = 1; msk < 16; msk <<= 1)
                sum += __shfl_xor(sum, msk, 32);
            if (c == 0) {
                const float sc = __expf(mold - mnew);  // exp(-inf)=0 on first block
                scale_s[row] = sc;
                l_s[row] = l_s[row] * sc + sum;
                m_s[row] = mnew;
            }
        }
        __syncthreads();

        // -------- phase 2: O = O*scale + P V  (this wave's 128-wide D slice)
        {
            float scv[8];
            #pragma unroll
            for (int r = 0; r < 8; ++r) scv[r] = scale_s[r + 8 * half];
            #pragma unroll
            for (int t = 0; t < 8; ++t)
                #pragma unroll
                for (int r = 0; r < 8; ++r) oacc[t][r] *= scv[r];

            #pragma unroll
            for (int kk2 = 0; kk2 < 128; kk2 += 32) {
                const v8h a0 = *(const v8h*)(sP + ln * SPW + kk2 + half * 16);
                const v8h a1 = *(const v8h*)(sP + ln * SPW + kk2 + half * 16 + 8);
                v16h a;
                #pragma unroll
                for (int e = 0; e < 8; ++e) { a[e] = a0[e]; a[8 + e] = a1[e]; }
                #pragma unroll
                for (int t = 0; t < 8; ++t) {
                    // B[k][n] = V[j0+kk2+k][d] = Vt[b][d][...]; contiguous 16 halfs
                    const _Float16* vcol = Vt + (size_t)bb * Dv * Nv
                                           + (size_t)(dbase + t * 16 + ln) * Nv
                                           + j0 + kk2 + half * 16;
                    const v16h b = *(const v16h*)vcol;
                    oacc[t] = wmma_f16(a, b, oacc[t]);
                }
            }
        }
        // no barrier needed: next phase1 touches sS/sQ only (disjoint from sP)
    }

    // -------- epilogue: divide by row sums, write f32 output --------------
    float linv[8];
    #pragma unroll
    for (int r = 0; r < 8; ++r) linv[r] = 1.0f / l_s[r + 8 * half];
    #pragma unroll
    for (int t = 0; t < 8; ++t)
        #pragma unroll
        for (int r = 0; r < 8; ++r) {
            const int row = q0 + r + 8 * half;
            const int col = dbase + t * 16 + ln;
            out[((size_t)bb * Nv + row) * Dv + col] = oacc[t][r] * linv[r];
        }
}

// ---------------------------------------------------------------------------
extern "C" void kernel_launch(void* const* d_in, const int* in_sizes, int n_in,
                              void* d_out, int out_size, void* d_ws, size_t ws_size,
                              hipStream_t stream) {
    const float* x  = (const float*)d_in[0];
    const float* Wq = (const float*)d_in[1];
    const float* Wk = (const float*)d_in[2];
    const float* Wv = (const float*)d_in[3];
    float* out = (float*)d_out;

    // Workspace: f16 Q (32MB) + f16 K (32MB) + f16 V-transposed (32MB)
    //          + f16 weights (6MB) = 102MB, all L2-resident (192MB).
    char* ws = (char*)d_ws;
    const size_t qkv_bytes = (size_t)Bv * Nv * Dv * sizeof(_Float16);
    _Float16* Qh = (_Float16*)(ws);
    _Float16* Kh = (_Float16*)(ws + qkv_bytes);
    _Float16* Vt = (_Float16*)(ws + 2 * qkv_bytes);
    _Float16* Wh = (_Float16*)(ws + 3 * qkv_bytes);

    dim3 g0((Dv * Dv) / (256 * 8), 3);
    convert_w_kernel<<<g0, dim3(256), 0, stream>>>(Wq, Wk, Wv, Wh);

    dim3 g1((Bv * Nv) / 64, Dv / 128, 3);
    qkv_gemm_kernel<<<g1, dim3(128), 0, stream>>>(x, Wh, Qh, Kh, Vt);

    dim3 g2(Nv / 16, Bv);
    attn_kernel<<<g2, dim3(256), 0, stream>>>(Qh, Kh, Vt, out);
}
